// GINLayer_13529146982749
// MI455X (gfx1250) — matrix operand order repaired and checked by reading, not compiled, not run on hardware
//
#include <hip/hip_runtime.h>

typedef __attribute__((ext_vector_type(16))) _Float16 v16h;
typedef __attribute__((ext_vector_type(8)))  _Float16 v8h;
typedef __attribute__((ext_vector_type(8)))  float    v8f;
typedef __attribute__((ext_vector_type(4)))  int      v4i;

// Address-space-qualified int4 for the gfx1250 async-to-LDS builtin:
// param 0 printed as 'int __vector(4) __device__ *'  -> AS1 (global)
// param 1 expected as the LDS destination            -> AS3 (shared)
typedef __attribute__((address_space(1))) v4i v4i_g;
typedef __attribute__((address_space(3))) v4i v4i_l;

#define N_NODES 100000
#define N_EDGES 800000
#define HID     128

#if __has_builtin(__builtin_amdgcn_global_load_async_to_lds_b128) && \
    __has_builtin(__builtin_amdgcn_s_wait_asynccnt)
#define HAVE_ASYNC_LDS 1
#else
#define HAVE_ASYNC_LDS 0
#endif

// ---------------------------------------------------------------------------
// Convert both weight matrices f32 -> f16 (each 128x128 = 16384 elements).
// ---------------------------------------------------------------------------
__global__ void gin_convert_weights(const float* __restrict__ W1,
                                    const float* __restrict__ W2,
                                    _Float16* __restrict__ W1h,
                                    _Float16* __restrict__ W2h) {
    int t = blockIdx.x * blockDim.x + threadIdx.x;   // 0 .. 32767
    if (t < HID * HID) {
        W1h[t] = (_Float16)W1[t];
    } else {
        int u = t - HID * HID;
        W2h[u] = (_Float16)W2[u];
    }
}

// ---------------------------------------------------------------------------
// h = x  (initializes the accumulator; fuses the "+ x" of GIN with eps=0)
// ---------------------------------------------------------------------------
__global__ void gin_init_h(const float* __restrict__ x, float* __restrict__ h) {
    int i = blockIdx.x * blockDim.x + threadIdx.x;   // over N*H/4 float4s
    ((float4*)h)[i] = ((const float4*)x)[i];
}

// ---------------------------------------------------------------------------
// Scatter-add: h[dst] += x[src].  One thread per (edge, 4-feature chunk):
// lanes within a wave cover consecutive feature chunks of the same edge,
// so the gather of x[src] is coalesced (512B row).  Atomics land in the
// 51MB accumulator which is L2-resident on the 192MB L2.
// ---------------------------------------------------------------------------
__global__ void gin_edge_scatter(const float* __restrict__ x,
                                 const int* __restrict__ edge_index,
                                 float* __restrict__ h) {
    int t = blockIdx.x * blockDim.x + threadIdx.x;   // E * 32 threads
    int e = t >> 5;           // edge id
    int c = t & 31;           // which float4 chunk of the 128 features
    if (e >= N_EDGES) return;
    int s = edge_index[e];             // row 0: src
    int d = edge_index[N_EDGES + e];   // row 1: dst
    float4 v = ((const float4*)(x + (size_t)s * HID))[c];
    float* p = h + (size_t)d * HID + c * 4;
    atomicAdd(p + 0, v.x);
    atomicAdd(p + 1, v.y);
    atomicAdd(p + 2, v.z);
    atomicAdd(p + 3, v.w);
}

// ---------------------------------------------------------------------------
// GEMM: Dst = act(A @ W^T + bias) using v_wmma_f32_16x16x32_f16.
//   A_F16        : A source is f16 (else f32, converted on the fly)
//   RELU_OUT_F16 : apply ReLU and store f16 (else store raw f32)
// Block = 256 threads = 8 waves; wave w handles rows [blk*128 + w*16, +16),
// producing all 128 output columns as 8 WMMA tiles, K consumed in 4 steps.
// W (f16, 32KB) is staged in LDS (async-to-LDS path) once per block.
// N_NODES is a multiple of 16, so each wave tile is fully in- or fully
// out-of-bounds: one uniform early-out, no per-element store predication.
// ---------------------------------------------------------------------------
template <bool A_F16, bool RELU_OUT_F16>
__global__ __launch_bounds__(256) void gin_gemm(
    const void* __restrict__ Asrc,        // [N, 128] f32 or f16, row-major
    const _Float16* __restrict__ W16,     // [128, 128] f16, row-major
    const float* __restrict__ bias,       // [128] f32
    void* __restrict__ Dst)               // [N, 128] f16 or f32
{
    __shared__ __align__(16) _Float16 sW[HID * HID];   // 32 KB

    // Stage weights into LDS: 2048 x 16B / 256 threads = 8 x b128 each.
    {
        int t = threadIdx.x;
#if HAVE_ASYNC_LDS
        char* g = (char*)W16 + t * 128;
        char* s = (char*)sW + t * 128;
#pragma unroll
        for (int i = 0; i < 8; ++i) {
            __builtin_amdgcn_global_load_async_to_lds_b128(
                (v4i_g*)(g + i * 16), (v4i_l*)(s + i * 16), 0, 0);
        }
        __builtin_amdgcn_s_wait_asynccnt(0);
#else
        const uint4* gg = (const uint4*)W16;
        uint4*       ss = (uint4*)sW;
#pragma unroll
        for (int i = 0; i < 8; ++i) ss[t * 8 + i] = gg[t * 8 + i];
#endif
    }
    __syncthreads();

    const int lane    = threadIdx.x & 31;
    const int wave    = threadIdx.x >> 5;
    const int rowbase = blockIdx.x * 128 + wave * 16;

    // Whole-tile bounds check (uniform per wave; after the barrier).
    if (rowbase >= N_NODES) return;

    const int m  = lane & 15;   // row within tile (A) / column (B,C,D)
    const int hi = lane >> 4;   // lane-half selector
    const int row = rowbase + m;

    // Preload A fragments for the whole K=128 (4 fragments of K=32).
    // A layout (matches B): lanes 0-15 hold K = [k0..k0+15];
    // lanes 16-31 hold the +16 half.  Two contiguous 8-half chunks per lane.
    v16h a[4];
#pragma unroll
    for (int ks = 0; ks < 4; ++ks) {
        const int k0 = ks * 32 + hi * 16;
        if (A_F16) {
            const _Float16* ap = (const _Float16*)Asrc + (size_t)row * HID + k0;
            v8h c0 = *(const v8h*)(ap);
            v8h c1 = *(const v8h*)(ap + 8);
#pragma unroll
            for (int e = 0; e < 8; ++e) { a[ks][e] = c0[e]; a[ks][8 + e] = c1[e]; }
        } else {
            const float* ap = (const float*)Asrc + (size_t)row * HID + k0;
            float4 f0 = *(const float4*)(ap);
            float4 f1 = *(const float4*)(ap + 4);
            float4 f2 = *(const float4*)(ap + 8);
            float4 f3 = *(const float4*)(ap + 12);
            a[ks][0]  = (_Float16)f0.x; a[ks][1]  = (_Float16)f0.y;
            a[ks][2]  = (_Float16)f0.z; a[ks][3]  = (_Float16)f0.w;
            a[ks][4]  = (_Float16)f1.x; a[ks][5]  = (_Float16)f1.y;
            a[ks][6]  = (_Float16)f1.z; a[ks][7]  = (_Float16)f1.w;
            a[ks][8]  = (_Float16)f2.x; a[ks][9]  = (_Float16)f2.y;
            a[ks][10] = (_Float16)f2.z; a[ks][11] = (_Float16)f2.w;
            a[ks][12] = (_Float16)f3.x; a[ks][13] = (_Float16)f3.y;
            a[ks][14] = (_Float16)f3.z; a[ks][15] = (_Float16)f3.w;
        }
    }

    // 8 output column tiles of 16.
    for (int j = 0; j < 8; ++j) {
        float bv = bias[j * 16 + m];
        v8f acc;
#pragma unroll
        for (int r = 0; r < 8; ++r) acc[r] = bv;

#pragma unroll
        for (int ks = 0; ks < 4; ++ks) {
            // B = W^T: lane n holds column n of B = row (j*16+n) of W,
            // 16 contiguous f16 at K-offset ks*32 (+16 for lanes 16-31).
            const _Float16* bp = sW + (size_t)(j * 16 + m) * HID + ks * 32 + hi * 16;
            v16h b = *(const v16h*)bp;
            acc = __builtin_amdgcn_wmma_f32_16x16x32_f16(
                false, a[ks], false, b, (short)0, acc, false, false);
        }

        // D layout: VGPR r -> M = r (lanes 0-15) or r+8 (lanes 16-31), N = m.
        const int col = j * 16 + m;
#pragma unroll
        for (int r = 0; r < 8; ++r) {
            const int orow = rowbase + r + hi * 8;
            if (RELU_OUT_F16) {
                float v = acc[r] > 0.0f ? acc[r] : 0.0f;
                ((_Float16*)Dst)[(size_t)orow * HID + col] = (_Float16)v;
            } else {
                ((float*)Dst)[(size_t)orow * HID + col] = acc[r];
            }
        }
    }
}

// ---------------------------------------------------------------------------
extern "C" void kernel_launch(void* const* d_in, const int* in_sizes, int n_in,
                              void* d_out, int out_size, void* d_ws, size_t ws_size,
                              hipStream_t stream) {
    const float* x  = (const float*)d_in[0];
    const int*   ei = (const int*)d_in[1];
    const float* W1 = (const float*)d_in[2];
    const float* b1 = (const float*)d_in[3];
    const float* W2 = (const float*)d_in[4];
    const float* b2 = (const float*)d_in[5];

    char* ws = (char*)d_ws;
    float*    h   = (float*)ws;                              // 51.2 MB f32 accumulator
    _Float16* h1  = (_Float16*)(ws + 51200000);              // 25.6 MB f16 hidden
    _Float16* w1h = (_Float16*)(ws + 76800000);              // 32 KB
    _Float16* w2h = (_Float16*)(ws + 76800000 + 32768);      // 32 KB

    // 1. Weights f32 -> f16
    gin_convert_weights<<<(2 * HID * HID) / 256, 256, 0, stream>>>(W1, W2, w1h, w2h);

    // 2. h = x
    gin_init_h<<<(N_NODES * HID / 4) / 256, 256, 0, stream>>>(x, h);

    // 3. h[dst] += x[src] over all edges (L2-resident f32 atomics)
    gin_edge_scatter<<<(N_EDGES * 32) / 256, 256, 0, stream>>>(x, ei, h);

    // 4. h1 = relu(h @ W1^T + b1), stored f16
    const int gemm_grid = (N_NODES + 127) / 128;
    gin_gemm<false, true><<<gemm_grid, 256, 0, stream>>>(h, w1h, b1, h1);

    // 5. out = h1 @ W2^T + b2, stored f32
    gin_gemm<true, false><<<gemm_grid, 256, 0, stream>>>(h1, w2h, b2, (float*)d_out);
}
